// WordContributionLearner_19224273617708
// MI455X (gfx1250) — compile-verified, matrix operand order checked by reading, not run on hardware
//
#include <hip/hip_runtime.h>
#include <hip/hip_bf16.h>
#include <math.h>

typedef __attribute__((ext_vector_type(16))) _Float16 v16h;
typedef __attribute__((ext_vector_type(8)))  float    v8f;

#define B_   64
#define D_   512
#define N_   50
#define NT_  64      // N padded to 4 tiles of 16
#define EH_  256     // E/2
#define HW_  1024

// ---------------------------------------------------------------------------
// Kernel A: swizzle attn_map_w (256x512 f32) into f16 WMMA A-fragments.
// Fragment (m_tile 0..15, k_step 0..15) = 512 halves, lane l, half j:
//   m = m_tile*16 + (l&15)
//   k = k_step*32 + (j>>3)*16 + (l>>4)*8 + (j&7)      (16-bit A 16x32 layout)
// ---------------------------------------------------------------------------
__global__ void swizzleA_kernel(const float* __restrict__ W, _Float16* __restrict__ wsA) {
    int idx    = blockIdx.x * blockDim.x + threadIdx.x;  // 0..131071
    int frag   = idx >> 9;
    int within = idx & 511;
    int l = within >> 4;
    int j = within & 15;
    int m_tile = frag >> 4;
    int k_step = frag & 15;
    int m = m_tile * 16 + (l & 15);
    int k = k_step * 32 + ((j >> 3) << 4) + ((l >> 4) << 3) + (j & 7);
    wsA[idx] = (_Float16)W[m * D_ + k];
}

// ---------------------------------------------------------------------------
// Kernel B: tile_visu[b,d] = mean_{h,w}( fvisu * bilinear4x(fvisu_ast) )
// One wave per (b,d); lane = w column -> coalesced fvisu reads.
// ---------------------------------------------------------------------------
__global__ void __launch_bounds__(256)
tile_visu_kernel(const float* __restrict__ fvisu,
                 const float* __restrict__ ast,
                 float* __restrict__ tv) {
    int b    = blockIdx.x >> 6;
    int dg   = blockIdx.x & 63;
    int wave = threadIdx.x >> 5;
    int lane = threadIdx.x & 31;
    int d    = dg * 8 + wave;

    __shared__ float astT[8][64];
    const float* ap = ast + ((size_t)(b * D_ + d)) * 64;
    astT[wave][lane]      = ap[lane];
    astT[wave][lane + 32] = ap[lane + 32];
    __syncthreads();

    // x interpolation (w == lane), half-pixel centers, edge clamp
    float sx  = 0.25f * (float)lane - 0.375f;
    float x0f = floorf(sx);
    float wx  = sx - x0f;
    int x0 = (int)x0f;
    int x1 = x0 + 1;
    x0 = x0 < 0 ? 0 : (x0 > 7 ? 7 : x0);
    x1 = x1 < 0 ? 0 : (x1 > 7 ? 7 : x1);

    const float* fp = fvisu + ((size_t)(b * D_ + d)) * HW_;
    float sum = 0.f;
    #pragma unroll 4
    for (int h = 0; h < 32; ++h) {
        float sy  = 0.25f * (float)h - 0.375f;
        float y0f = floorf(sy);
        float wy  = sy - y0f;
        int y0 = (int)y0f;
        int y1 = y0 + 1;
        y0 = y0 < 0 ? 0 : (y0 > 7 ? 7 : y0);
        y1 = y1 < 0 ? 0 : (y1 > 7 ? 7 : y1);
        float a00 = astT[wave][y0 * 8 + x0];
        float a01 = astT[wave][y0 * 8 + x1];
        float a10 = astT[wave][y1 * 8 + x0];
        float a11 = astT[wave][y1 * 8 + x1];
        float wa  = (1.f - wy) * ((1.f - wx) * a00 + wx * a01)
                  +        wy  * ((1.f - wx) * a10 + wx * a11);
        sum += fp[h * 32 + lane] * wa;
    }
    #pragma unroll
    for (int off = 16; off > 0; off >>= 1)
        sum += __shfl_down(sum, off, 32);
    if (lane == 0) tv[b * D_ + d] = sum * (1.f / 1024.f);
}

// ---------------------------------------------------------------------------
// Kernel C: per-batch GEMM (256x512 @ 512x64) via v_wmma_f32_16x16x32_f16,
// then score[b,n] = sum_e score_w[e]*tanh(cs[n]*dot[e,n] + map_b[e]) + score_b
// One block = one batch; 8 waves; wave owns 2 m-tiles x 4 n-tiles.
// B-fragments of g[d,n] = tv[d]*fword[n,d] staged in LDS (2 K-phases, 32 KB).
// ---------------------------------------------------------------------------
__global__ void __launch_bounds__(256)
gemm_score_kernel(const _Float16* __restrict__ wsA,
                  const float* __restrict__ tv,
                  const float* __restrict__ fword,
                  const float* __restrict__ cscore,
                  const float* __restrict__ map_b,
                  const float* __restrict__ score_w,
                  const float* __restrict__ score_b,
                  float* __restrict__ out_score) {
    int b    = blockIdx.x;
    int tid  = threadIdx.x;
    int wave = tid >> 5;
    int lane = tid & 31;

    __shared__ __attribute__((aligned(32))) _Float16 gfrag[4 * 8 * 512]; // 32 KB
    __shared__ float score_lds[64];
    if (tid < 64) score_lds[tid] = 0.f;

    const float* tvb = tv + b * D_;
    const float* fwb = fword + (size_t)b * N_ * D_;

    v8f acc[2][4] = {};

    for (int p = 0; p < 2; ++p) {
        __syncthreads();
        // Build B-fragments for k_steps p*8 .. p*8+7 (all 4 n-tiles).
        for (int idx = tid; idx < 4 * 8 * 512; idx += 256) {
            int frag   = idx >> 9;      // [nt 0..3][ks 0..7]
            int nt     = frag >> 3;
            int ks     = frag & 7;
            int within = idx & 511;
            int l = within >> 4;
            int j = within & 15;
            int n  = nt * 16 + (l & 15);
            int kl = ((j >> 3) << 4) + ((l >> 4) << 3) + (j & 7);
            int d  = (p * 8 + ks) * 32 + kl;
            float v = (n < N_) ? tvb[d] * fwb[n * D_ + d] : 0.f;
            gfrag[idx] = (_Float16)v;
        }
        __syncthreads();

        #pragma unroll
        for (int mt = 0; mt < 2; ++mt) {
            int m_tile = wave * 2 + mt;
            for (int ks = 0; ks < 8; ++ks) {
                int k_step = p * 8 + ks;
                v16h a = *(const v16h*)(wsA + ((size_t)(m_tile * 16 + k_step)) * 512
                                            + lane * 16);
                #pragma unroll
                for (int nt = 0; nt < 4; ++nt) {
                    v16h bb = *(const v16h*)(&gfrag[(nt * 8 + ks) * 512 + lane * 16]);
                    acc[mt][nt] = __builtin_amdgcn_wmma_f32_16x16x32_f16(
                        false, a, false, bb, (short)0, acc[mt][nt], false, false);
                }
            }
        }
    }
    __syncthreads();

    // Epilogue: C/D layout -> lane holds n = nt*16+(lane&15),
    // VGPR v holds e = m_tile*16 + (lane>=16 ? 8 : 0) + v.
    #pragma unroll
    for (int nt = 0; nt < 4; ++nt) {
        int n      = nt * 16 + (lane & 15);
        bool valid = (n < N_);
        float cs   = valid ? cscore[b * N_ + n] : 0.f;
        float partial = 0.f;
        #pragma unroll
        for (int mt = 0; mt < 2; ++mt) {
            int ebase = (wave * 2 + mt) * 16 + ((lane >> 4) << 3);
            #pragma unroll
            for (int v = 0; v < 8; ++v) {
                int e   = ebase + v;
                float t = tanhf(cs * acc[mt][nt][v] + map_b[e]);
                partial += score_w[e] * t;
            }
        }
        if (valid) atomicAdd(&score_lds[n], partial);
    }
    __syncthreads();
    if (tid < N_) out_score[b * N_ + tid] = score_lds[tid] + score_b[0];
}

// ---------------------------------------------------------------------------
// Kernel D: masked softmax (temp=10, zero idx 0 and idx s-1, renorm) and
// attn_lang[b,d] = sum_n ms[n] * fword[b,n,d]
// ---------------------------------------------------------------------------
__global__ void __launch_bounds__(256)
softmax_lang_kernel(const float* __restrict__ fword,
                    const int* __restrict__ wmask,
                    const float* __restrict__ score_in,
                    float* __restrict__ out_lang) {
    int b   = blockIdx.x;
    int tid = threadIdx.x;
    __shared__ float ms[N_];

    if (tid == 0) {
        float e[N_];
        int s = 0;
        float mx = -1e30f;
        for (int n = 0; n < N_; ++n) {
            s += wmask[b * N_ + n];
            float v = 10.f * score_in[b * N_ + n];
            e[n] = v;
            mx = fmaxf(mx, v);
        }
        float sum = 0.f;
        for (int n = 0; n < N_; ++n) { e[n] = expf(e[n] - mx); sum += e[n]; }
        float inv_sum = 1.f / sum;
        float msum = 0.f;
        for (int n = 0; n < N_; ++n) {
            float pnorm = e[n] * inv_sum;
            float m = (wmask[b * N_ + n] != 0 && n != 0 && n != (s - 1)) ? pnorm : 0.f;
            ms[n] = m;
            msum += m;
        }
        float inv = 1.f / (msum + 1e-8f);
        for (int n = 0; n < N_; ++n) ms[n] *= inv;
    }
    __syncthreads();

    const float* fwb = fword + (size_t)b * N_ * D_;
    for (int d = tid; d < D_; d += 256) {
        float a = 0.f;
        for (int n = 0; n < N_; ++n) a += ms[n] * fwb[n * D_ + d];
        out_lang[b * D_ + d] = a;
    }
}

// ---------------------------------------------------------------------------
extern "C" void kernel_launch(void* const* d_in, const int* in_sizes, int n_in,
                              void* d_out, int out_size, void* d_ws, size_t ws_size,
                              hipStream_t stream) {
    const float* fvisu  = (const float*)d_in[0];
    const float* fword  = (const float*)d_in[1];
    const float* ast    = (const float*)d_in[2];
    const float* cscore = (const float*)d_in[3];
    const int*   wmask  = (const int*)  d_in[4];
    const float* map_w  = (const float*)d_in[5];
    const float* map_b  = (const float*)d_in[6];
    const float* sc_w   = (const float*)d_in[7];
    const float* sc_b   = (const float*)d_in[8];

    float* out_lang  = (float*)d_out;            // [64,512]
    float* out_score = (float*)d_out + B_ * D_;  // [64,50]

    _Float16* wsA = (_Float16*)d_ws;                       // 131072 halves = 256 KB
    float*    tv  = (float*)((char*)d_ws + 131072 * 2);    // 32768 f32    = 128 KB

    swizzleA_kernel<<<512, 256, 0, stream>>>(map_w, wsA);
    tile_visu_kernel<<<B_ * 64, 256, 0, stream>>>(fvisu, ast, tv);
    gemm_score_kernel<<<B_, 256, 0, stream>>>(wsA, tv, fword, cscore,
                                              map_b, sc_w, sc_b, out_score);
    softmax_lang_kernel<<<B_, 256, 0, stream>>>(fword, wmask, out_score, out_lang);
}
